// ClothGraphConvNetwork_MLPDecoder_74045236183238
// MI455X (gfx1250) — matrix operand order repaired and checked by reading, not compiled, not run on hardware
//
#include <hip/hip_runtime.h>
#include <hip/hip_bf16.h>

typedef __attribute__((ext_vector_type(16))) __bf16 v16bf;
typedef __attribute__((ext_vector_type(8)))  float  v8f;

union BF16x16 { v16bf v; uint4 q[2]; };

static __device__ __forceinline__ unsigned short f2bf(float f) {
  unsigned int u = __float_as_uint(f);
  unsigned int r = (u + 0x7FFFu + ((u >> 16) & 1u)) >> 16;   // RNE
  return (unsigned short)r;
}
static __device__ __forceinline__ float bf2f(unsigned short h) {
  return __uint_as_float((unsigned int)h << 16);
}

// ---------------------------------------------------------------------------
// GEMM: Y(M,O) = X(M,K) @ W^T ; X,W in bf16 (W pre-converted to (O,K)).
// flags: 1 = accumulate into Y (fp32 residual), 2 = ReLU, 8 = store bf16
// Tile 128(M) x 64(O) per 256-thread WG; each wave owns 32x32 -> 4 WMMA/K-step.
// ---------------------------------------------------------------------------
__global__ __launch_bounds__(256)
void gemm_bf16_wmma(const unsigned short* __restrict__ X,
                    const unsigned short* __restrict__ W,
                    const float* __restrict__ bias, void* __restrict__ Yv,
                    int M, int K, int O, int flags) {
  __shared__ unsigned short As[128 * 32];   // 8 KB
  __shared__ unsigned short Bs[64 * 32];    // 4 KB
  const int tid  = threadIdx.x;
  const int lane = tid & 31;
  const int wid  = tid >> 5;                // 8 waves (wave32)
  const int wm   = wid >> 1;                // 0..3  -> 32-row strip
  const int wo   = wid & 1;                 // 0..1  -> 32-col strip
  const int half = lane >> 4;
  const int ml   = lane & 15;
  const int m0   = blockIdx.x * 128;
  const int o0   = blockIdx.y * 64;

  v8f acc[2][2];
  for (int i = 0; i < 2; ++i)
    for (int j = 0; j < 2; ++j)
      for (int v = 0; v < 8; ++v) acc[i][j][v] = 0.f;

  for (int k0 = 0; k0 < K; k0 += 32) {
    // ---- stage X tile (128x32 bf16): 512 uint4 chunks, 2 per thread ----
    for (int i = 0; i < 2; ++i) {
      int idx = tid + i * 256;              // 0..511
      int r   = idx >> 2;
      int kc  = (idx & 3) << 3;             // 0,8,16,24
      uint4 v; v.x = v.y = v.z = v.w = 0u;
      int m = m0 + r;
      if (m < M)
        v = *reinterpret_cast<const uint4*>(X + (size_t)m * K + k0 + kc);
      *reinterpret_cast<uint4*>(&As[r * 32 + kc]) = v;
    }
    // ---- stage W tile (64x32 bf16): 256 uint4 chunks, 1 per thread ----
    {
      int r  = tid >> 2;                    // output col 0..63
      int kc = (tid & 3) << 3;
      uint4 v; v.x = v.y = v.z = v.w = 0u;
      if (o0 + r < O)
        v = *reinterpret_cast<const uint4*>(W + (size_t)(o0 + r) * K + k0 + kc);
      *reinterpret_cast<uint4*>(&Bs[r * 32 + kc]) = v;
    }
    __syncthreads();

    // ---- fragments per documented wave32 VGPR layouts ----
    BF16x16 a[2], b[2];
    for (int f = 0; f < 2; ++f) {
      const int ar = (wm * 32 + f * 16 + ml) * 32;
      a[f].q[0] = *reinterpret_cast<const uint4*>(&As[ar + half * 8]);
      a[f].q[1] = *reinterpret_cast<const uint4*>(&As[ar + 16 + half * 8]);
      const int br = (wo * 32 + f * 16 + ml) * 32;
      b[f].q[0] = *reinterpret_cast<const uint4*>(&Bs[br + half * 8]);
      b[f].q[1] = *reinterpret_cast<const uint4*>(&Bs[br + 16 + half * 8]);
    }
    acc[0][0] = __builtin_amdgcn_wmma_f32_16x16x32_bf16(false, a[0].v, false, b[0].v,
                                                        (short)0, acc[0][0], false, false);
    acc[0][1] = __builtin_amdgcn_wmma_f32_16x16x32_bf16(false, a[0].v, false, b[1].v,
                                                        (short)0, acc[0][1], false, false);
    acc[1][0] = __builtin_amdgcn_wmma_f32_16x16x32_bf16(false, a[1].v, false, b[0].v,
                                                        (short)0, acc[1][0], false, false);
    acc[1][1] = __builtin_amdgcn_wmma_f32_16x16x32_bf16(false, a[1].v, false, b[1].v,
                                                        (short)0, acc[1][1], false, false);
    __syncthreads();
  }

  // ---- epilogue: bias (+residual) (+ReLU), fp32 or bf16 store ----
  for (int i = 0; i < 2; ++i)
    for (int j = 0; j < 2; ++j)
      for (int v = 0; v < 8; ++v) {
        int row = m0 + wm * 32 + i * 16 + half * 8 + v;
        int col = o0 + wo * 32 + j * 16 + ml;
        if (row < M && col < O) {
          float r = acc[i][j][v] + (bias ? bias[col] : 0.f);
          if (flags & 1) r += reinterpret_cast<float*>(Yv)[(size_t)row * O + col];
          if (flags & 2) r = fmaxf(r, 0.f);
          if (flags & 8)
            reinterpret_cast<unsigned short*>(Yv)[(size_t)row * O + col] = f2bf(r);
          else
            reinterpret_cast<float*>(Yv)[(size_t)row * O + col] = r;
        }
      }
}

// ---------------------------------------------------------------------------
// GroupNorm (+optional ReLU). Group = 8 contiguous channels, layout (B*N, C).
// Input fp32 or bf16 (flags&2), output always bf16 (feeds WMMA GEMMs).
// ---------------------------------------------------------------------------
__global__ __launch_bounds__(256)
void gn_relu_kernel(const void* __restrict__ xv, unsigned short* __restrict__ y,
                    const float* __restrict__ g, const float* __restrict__ bt,
                    int Nv, int Cc, int flags /*1=relu, 2=in_bf16*/) {
  const int G  = Cc >> 3;
  const int bb = blockIdx.x / G;
  const int gg = blockIdx.x % G;
  const size_t base = (size_t)bb * Nv * Cc + gg * 8;
  const float* xf = (const float*)xv;
  const unsigned short* xh = (const unsigned short*)xv;
  const int tot = Nv * 8;

  float s = 0.f, q = 0.f;
  for (int idx = threadIdx.x; idx < tot; idx += 256) {
    int n = idx >> 3, c = idx & 7;
    size_t off = base + (size_t)n * Cc + c;
    float v = (flags & 2) ? bf2f(xh[off]) : xf[off];
    s += v; q += v * v;
  }
  __shared__ float ss[256], sq[256];
  ss[threadIdx.x] = s; sq[threadIdx.x] = q;
  __syncthreads();
  for (int st = 128; st > 0; st >>= 1) {
    if (threadIdx.x < st) { ss[threadIdx.x] += ss[threadIdx.x + st];
                            sq[threadIdx.x] += sq[threadIdx.x + st]; }
    __syncthreads();
  }
  float mean = ss[0] / tot;
  float var  = sq[0] / tot - mean * mean;
  float rs   = rsqrtf(var + 1e-5f);

  for (int idx = threadIdx.x; idx < tot; idx += 256) {
    int n = idx >> 3, c = idx & 7;
    size_t off = base + (size_t)n * Cc + c;
    float v = (flags & 2) ? bf2f(xh[off]) : xf[off];
    v = (v - mean) * rs * g[gg * 8 + c] + bt[gg * 8 + c];
    if (flags & 1) v = fmaxf(v, 0.f);
    y[off] = f2bf(v);
  }
}

// ---------------------------------------------------------------------------
// SpMM gather: dst = repeat(arange(N),8) => row n owns edges [8n,8n+8). fp32.
// ---------------------------------------------------------------------------
__global__ __launch_bounds__(256)
void spmm_kernel(const float* __restrict__ sup, const int* __restrict__ src,
                 const float* __restrict__ w, const float* __restrict__ cb,
                 float* __restrict__ out, int Nv) {
  const int mn = blockIdx.x;            // b*Nv + n
  const int n  = mn % Nv;
  const int bb = mn / Nv;
  const int c  = threadIdx.x;           // 256 channels
  const int e0 = n * 8;
  const float* base = sup + (size_t)bb * Nv * 256;
  float acc = cb[c];
  for (int j = 0; j < 8; ++j)
    acc += w[e0 + j] * base[(size_t)src[e0 + j] * 256 + c];
  out[(size_t)mn * 256 + c] = acc;
}

// ---------------------------------------------------------------------------
// lin0 algebraic split: image part broadcast over N -> per-batch projection.
// ---------------------------------------------------------------------------
__global__ __launch_bounds__(256)
void imgproj_kernel(const float* __restrict__ img, const float* __restrict__ W0,
                    float* __restrict__ ip) {
  int idx = blockIdx.x * 256 + threadIdx.x;     // B*1024
  if (idx >= 16 * 1024) return;
  int b = idx >> 10, o = idx & 1023;
  const float* wr = W0 + (size_t)o * 2051 + 3;
  const float* im = img + (size_t)b * 2048;
  float acc = 0.f;
  for (int i = 0; i < 2048; ++i) acc += wr[i] * im[i];
  ip[idx] = acc;
}

__global__ __launch_bounds__(256)
void assemble_kernel(const float* __restrict__ ip, const float* __restrict__ W0,
                     const float* __restrict__ b0, const float* __restrict__ verts,
                     unsigned short* __restrict__ x0, int Nv, int M) {
  size_t idx = (size_t)blockIdx.x * 256 + threadIdx.x;   // M*1024
  if (idx >= (size_t)M * 1024) return;
  int o = (int)(idx & 1023);
  int m = (int)(idx >> 10);
  int b = m / Nv, n = m % Nv;
  const float* wr = W0 + (size_t)o * 2051;
  float v = ip[(b << 10) + o] + b0[o]
          + wr[0] * verts[n] + wr[1] * verts[Nv + n] + wr[2] * verts[2 * Nv + n];
  x0[idx] = f2bf(v);
}

// ---------------------------------------------------------------------------
// fp32 -> bf16 converters (weights: once per launch; residual stream: head)
// ---------------------------------------------------------------------------
__global__ __launch_bounds__(256)
void cvt_bf16_kernel(const float* __restrict__ x, unsigned short* __restrict__ y,
                     size_t n) {
  size_t i = (size_t)blockIdx.x * 256 + threadIdx.x;
  if (i < n) y[i] = f2bf(x[i]);
}
// x is rows x cols; y is cols x rows (transpose): for conv weights (K,O)->(O,K)
__global__ __launch_bounds__(256)
void cvt_bf16_t_kernel(const float* __restrict__ x, unsigned short* __restrict__ y,
                       int rows, int cols) {
  int i = blockIdx.x * 256 + threadIdx.x;
  if (i >= rows * cols) return;
  int r = i / cols, c = i % cols;
  y[(size_t)c * rows + r] = f2bf(x[i]);
}

// ---------------------------------------------------------------------------
// head h3: (3,32) @ t(bf16) -> out (B,3,N) fp32
// ---------------------------------------------------------------------------
__global__ __launch_bounds__(256)
void head3_kernel(const unsigned short* __restrict__ t, const float* __restrict__ W,
                  const float* __restrict__ bsc, float* __restrict__ out,
                  int Nv, int M) {
  int m = blockIdx.x * 256 + threadIdx.x;
  if (m >= M) return;
  int b = m / Nv, n = m % Nv;
  const unsigned short* tr = t + (size_t)m * 32;
  for (int o = 0; o < 3; ++o) {
    float acc = bsc[o];
    for (int c = 0; c < 32; ++c) acc += W[o * 32 + c] * bf2f(tr[c]);
    out[((size_t)b * 3 + o) * Nv + n] = acc;
  }
}

// ---------------------------------------------------------------------------
static inline void gemm(const unsigned short* X, const unsigned short* W,
                        const float* bias, void* Y,
                        int M, int K, int O, int flags, hipStream_t s) {
  dim3 g((M + 127) / 128, (O + 63) / 64);
  gemm_bf16_wmma<<<g, dim3(256), 0, s>>>(X, W, bias, Y, M, K, O, flags);
}
static inline void cvt(const float* x, unsigned short* y, size_t n, hipStream_t s) {
  cvt_bf16_kernel<<<(int)((n + 255) / 256), 256, 0, s>>>(x, y, n);
}

extern "C" void kernel_launch(void* const* d_in, const int* in_sizes, int n_in,
                              void* d_out, int out_size, void* d_ws, size_t ws_size,
                              hipStream_t stream) {
  const int Bn = 16, Nv = 1723, M = Bn * Nv;   // 27568

  const float* image     = (const float*)d_in[0];
  const float* verts     = (const float*)d_in[1];
  const int*   adj_src   = (const int*)  d_in[2];
  const float* adj_w     = (const float*)d_in[4];
  const float* lin0_W    = (const float*)d_in[5];
  const float* lin0_b    = (const float*)d_in[6];
  const float* b0_pre_g  = (const float*)d_in[7];
  const float* b0_pre_b  = (const float*)d_in[8];
  const float* b0_lin1_W = (const float*)d_in[9];
  const float* b0_lin1_b = (const float*)d_in[10];
  const float* b0_n1_g   = (const float*)d_in[11];
  const float* b0_n1_b   = (const float*)d_in[12];
  const float* b0_conv_W = (const float*)d_in[13];
  const float* b0_conv_b = (const float*)d_in[14];
  const float* b0_n2_g   = (const float*)d_in[15];
  const float* b0_n2_b   = (const float*)d_in[16];
  const float* b0_lin2_W = (const float*)d_in[17];
  const float* b0_lin2_b = (const float*)d_in[18];
  const float* b0_skip_W = (const float*)d_in[19];
  const float* b0_skip_b = (const float*)d_in[20];
  const float* blk_pre_g  = (const float*)d_in[21];
  const float* blk_pre_b  = (const float*)d_in[22];
  const float* blk_lin1_W = (const float*)d_in[23];
  const float* blk_lin1_b = (const float*)d_in[24];
  const float* blk_n1_g   = (const float*)d_in[25];
  const float* blk_n1_b   = (const float*)d_in[26];
  const float* blk_conv_W = (const float*)d_in[27];
  const float* blk_conv_b = (const float*)d_in[28];
  const float* blk_n2_g   = (const float*)d_in[29];
  const float* blk_n2_b   = (const float*)d_in[30];
  const float* blk_lin2_W = (const float*)d_in[31];
  const float* blk_lin2_b = (const float*)d_in[32];
  const float* h1_W = (const float*)d_in[33];
  const float* h1_b = (const float*)d_in[34];
  const float* h2_W = (const float*)d_in[35];
  const float* h2_b = (const float*)d_in[36];
  const float* hn_g = (const float*)d_in[37];
  const float* hn_b = (const float*)d_in[38];
  const float* h3_W = (const float*)d_in[39];
  const float* h3_b = (const float*)d_in[40];
  float* out = (float*)d_out;

  // ---- workspace (byte allocator, 256B aligned regions) ----
  char* p = (char*)d_ws;
  auto alloc = [&](size_t bytes) { char* r = p; p += (bytes + 255) & ~(size_t)255; return r; };
  float*          Xb  = (float*)         alloc((size_t)M * 512 * 4);   // residual (fp32)
  float*          c1f = (float*)         alloc((size_t)M * 256 * 4);
  float*          c2f = (float*)         alloc((size_t)M * 256 * 4);
  unsigned short* gnb = (unsigned short*)alloc((size_t)M * 1024 * 2);  // GN out / Xb-bf16
  unsigned short* x0h = (unsigned short*)alloc((size_t)M * 1024 * 2);  // lin0 out (bf16)
  unsigned short* c1h = (unsigned short*)alloc((size_t)M * 256 * 2);
  float*          ip  = (float*)         alloc((size_t)Bn * 1024 * 4);
  unsigned short* wp  = (unsigned short*)alloc((size_t)2656256 * 2);   // bf16 weights

  // bf16 weight pool layout
  unsigned short* w_b0l1 = wp;                      // 256*1024
  unsigned short* w_b0cv = w_b0l1 + 262144;         // 256*256 (transposed)
  unsigned short* w_b0l2 = w_b0cv + 65536;          // 512*256
  unsigned short* w_b0sk = w_b0l2 + 131072;         // 512*1024
  unsigned short* w_bl1  = w_b0sk + 524288;         // 5*256*512
  unsigned short* w_bcv  = w_bl1  + 655360;         // 5*256*256 (transposed)
  unsigned short* w_bl2  = w_bcv  + 327680;         // 5*512*256
  unsigned short* w_h1   = w_bl2  + 655360;         // 64*512
  unsigned short* w_h2   = w_h1   + 32768;          // 32*64

  // ---- one-shot weight conversion (fp32 -> bf16, conv transposed to (O,K)) ----
  cvt(b0_lin1_W, w_b0l1, 262144, stream);
  cvt_bf16_t_kernel<<<(65536 + 255) / 256, 256, 0, stream>>>(b0_conv_W, w_b0cv, 256, 256);
  cvt(b0_lin2_W, w_b0l2, 131072, stream);
  cvt(b0_skip_W, w_b0sk, 524288, stream);
  cvt(blk_lin1_W, w_bl1, 655360, stream);
  for (int i = 0; i < 5; ++i)
    cvt_bf16_t_kernel<<<(65536 + 255) / 256, 256, 0, stream>>>(
        blk_conv_W + (size_t)i * 65536, w_bcv + (size_t)i * 65536, 256, 256);
  cvt(blk_lin2_W, w_bl2, 655360, stream);
  cvt(h1_W, w_h1, 32768, stream);
  cvt(h2_W, w_h2, 2048, stream);

  // ---- lin0 (algebraic split; output bf16) ----
  imgproj_kernel<<<(Bn * 1024 + 255) / 256, 256, 0, stream>>>(image, lin0_W, ip);
  assemble_kernel<<<(int)(((size_t)M * 1024 + 255) / 256), 256, 0, stream>>>(
      ip, lin0_W, lin0_b, verts, x0h, Nv, M);

  // ---- b0 resblock (1024 -> 256 -> 512, with skip GEMM) ----
  gn_relu_kernel<<<Bn * 128, 256, 0, stream>>>(x0h, gnb, b0_pre_g, b0_pre_b, Nv, 1024, 1 | 2);
  gemm(gnb, w_b0l1, b0_lin1_b, c1f, M, 1024, 256, 0, stream);
  gn_relu_kernel<<<Bn * 32, 256, 0, stream>>>(c1f, c1h, b0_n1_g, b0_n1_b, Nv, 256, 1);
  gemm(c1h, w_b0cv, nullptr, c2f, M, 256, 256, 0, stream);          // sup
  spmm_kernel<<<M, 256, 0, stream>>>(c2f, adj_src, adj_w, b0_conv_b, c1f, Nv);
  gn_relu_kernel<<<Bn * 32, 256, 0, stream>>>(c1f, c1h, b0_n2_g, b0_n2_b, Nv, 256, 1);
  gemm(x0h, w_b0sk, b0_skip_b, Xb, M, 1024, 512, 0, stream);        // skip path
  gemm(c1h, w_b0l2, b0_lin2_b, Xb, M, 256, 512, 1, stream);         // X += y

  // ---- 5 resblocks (512 -> 256 -> 512, identity skip fused as accumulate) ----
  for (int i = 0; i < 5; ++i) {
    gn_relu_kernel<<<Bn * 64, 256, 0, stream>>>(Xb, gnb, blk_pre_g + i * 512,
                                                blk_pre_b + i * 512, Nv, 512, 1);
    gemm(gnb, w_bl1 + (size_t)i * 131072, blk_lin1_b + i * 256, c1f, M, 512, 256, 0, stream);
    gn_relu_kernel<<<Bn * 32, 256, 0, stream>>>(c1f, c1h, blk_n1_g + i * 256,
                                                blk_n1_b + i * 256, Nv, 256, 1);
    gemm(c1h, w_bcv + (size_t)i * 65536, nullptr, c2f, M, 256, 256, 0, stream);
    spmm_kernel<<<M, 256, 0, stream>>>(c2f, adj_src, adj_w, blk_conv_b + i * 256, c1f, Nv);
    gn_relu_kernel<<<Bn * 32, 256, 0, stream>>>(c1f, c1h, blk_n2_g + i * 256,
                                                blk_n2_b + i * 256, Nv, 256, 1);
    gemm(c1h, w_bl2 + (size_t)i * 131072, blk_lin2_b + i * 512, Xb, M, 256, 512, 1, stream);
  }

  // ---- decoder head ----
  cvt(Xb, gnb, (size_t)M * 512, stream);                            // residual -> bf16
  gemm(gnb, w_h1, h1_b, c1h, M, 512, 64, 2 | 8, stream);            // ReLU, bf16 out
  gemm(c1h, w_h2, h2_b, c2f, M, 64, 32, 0, stream);
  gn_relu_kernel<<<Bn * 4, 256, 0, stream>>>(c2f, c1h, hn_g, hn_b, Nv, 32, 1);
  head3_kernel<<<(M + 255) / 256, 256, 0, stream>>>(c1h, h3_W, h3_b, out, Nv, M);
}
